// VanillaRNN_20418274525461
// MI455X (gfx1250) — matrix operand order repaired and checked by reading, not compiled, not run on hardware
//
#include <hip/hip_runtime.h>
#include <stdint.h>

// Problem constants (from reference): V=128, E=256, H=512, B=128, S=2048
#define B_   128
#define S_   2048
#define V_   128
#define E_   256
#define H_   512
#define HROW 520   // padded LDS row stride (bf16 elements): 520*2B = 1040B -> 4-bank skew

typedef __attribute__((ext_vector_type(16))) __bf16 v16bf;
typedef __attribute__((ext_vector_type(8)))  float  v8f;
typedef __attribute__((ext_vector_type(4)))  unsigned int v4u;
typedef __attribute__((ext_vector_type(8)))  int    v8i;
typedef __attribute__((ext_vector_type(4)))  int    v4i;

union BF16Frag { v16bf v; uint4 q[2]; };
union F8       { v8f   v; float f[8]; };

__device__ __forceinline__ unsigned short f2bf(float x) {
  union { float f; unsigned u; } v; v.f = x;
  unsigned r = v.u + 0x7FFFu + ((v.u >> 16) & 1u);   // round-to-nearest-even
  return (unsigned short)(r >> 16);
}
__device__ __forceinline__ float bf2f(unsigned short h) {
  union { unsigned u; float f; } v; v.u = ((unsigned)h) << 16;
  return v.f;
}

#if __has_builtin(__builtin_amdgcn_tensor_load_to_lds) && \
    __has_builtin(__builtin_amdgcn_s_wait_tensorcnt)
#define HAVE_TDM 1
#else
#define HAVE_TDM 0
#endif

#if HAVE_TDM
// Issue a TDM load of a 16x512 bf16 tile (row stride 512 elements in memory)
// into LDS at byte offset lds_off, padding each 1024B row by 16B -> HROW=520.
// D# layout per CDNA5 ISA (2D tensor, groups 2/3 zero).
// This toolchain exposes the 6-arg builtin form (extra zero-filled int32x8 group).
__device__ __forceinline__ void tdm_load_tile(unsigned lds_off,
                                              const unsigned short* gsrc) {
  unsigned long long ga = (unsigned long long)(uintptr_t)gsrc;
  v4u g0;
  g0.x = 1u;                                        // count=1 (valid user D#)
  g0.y = lds_off;                                   // lds_addr (bytes)
  g0.z = (unsigned)(ga & 0xFFFFFFFFu);              // global_addr[31:0]
  g0.w = (unsigned)((ga >> 32) & 0x01FFFFFFu)       // global_addr[56:32]
       | (2u << 30);                                // type = 2 ("image")
  v8i g1;
  g1[0] = (1 << 16)     // data_size = 1 -> 2 bytes
        | (1 << 20)     // pad_enable (LDS row skew)
        | (7 << 22)     // pad_interval = 256 DWORDs (one 1024B row)
        | (3 << 25);    // pad_amount   = 4 DWORDs (16B -> 520-elem row stride)
  g1[1] = (int)(512u << 16);  // tensor_dim0[15:0]=512 at bits[63:48]
  g1[2] = (int)(16u  << 16);  // tensor_dim0 hi=0; tensor_dim1[15:0]=16 at [95:80]
  g1[3] = (int)(512u << 16);  // tensor_dim1 hi=0; tile_dim0=512 at [127:112]
  g1[4] = 16;                 // tile_dim1=16 at [143:128]; tile_dim2=0
  g1[5] = 512;                // tensor_dim0_stride low 32 = 512 elements
  g1[6] = 0;                  // stride hi / tensor_dim1_stride lo = 0
  g1[7] = 0;
  v4i z4 = {0, 0, 0, 0};
  v8i z8 = {0, 0, 0, 0, 0, 0, 0, 0};
  __builtin_amdgcn_tensor_load_to_lds(g0, g1, z4, z4, z8, 0);
}
#endif

// ---------------- tiny prep kernels: fp32 -> bf16 (and transpose) ----------------
__global__ void cvt_bf16_kernel(const float* __restrict__ src,
                                unsigned short* __restrict__ dst, int n) {
  int i = blockIdx.x * blockDim.x + threadIdx.x;
  if (i < n) dst[i] = f2bf(src[i]);
}

// dst[c][r] = src[r][c]  (src: rows x cols fp32, dst: cols x rows bf16)
__global__ void transpose_bf16_kernel(const float* __restrict__ src,
                                      unsigned short* __restrict__ dst,
                                      int rows, int cols) {
  int i = blockIdx.x * blockDim.x + threadIdx.x;
  if (i < rows * cols) {
    int r = i / cols, c = i - r * cols;
    dst[(size_t)c * rows + r] = f2bf(src[i]);
  }
}

// ---------------- phase 1: xw[t][b][h] = emb[x[b,t]] @ Wx + bx (bf16 out) -------
// grid: (S_, H_/64); block: 256 = 8 waves; wave w handles batch rows w*16..w*16+15,
// 64 columns (4 WMMA accum tiles). A gathered from emb rows, B from Wx^T (col-major).
__global__ __launch_bounds__(256) void rnn_xw_kernel(
    const int* __restrict__ x, const unsigned short* __restrict__ embb,
    const unsigned short* __restrict__ wxT, const float* __restrict__ bx,
    unsigned short* __restrict__ xw)
{
  const int t    = blockIdx.x;
  const int n0   = blockIdx.y * 64;
  const int wave = threadIdx.x >> 5;
  const int lane = threadIdx.x & 31;
  const int hl   = lane >> 4;       // lane half
  const int ln   = lane & 15;
  const int aklo = hl ? 8 : 0;      // A frag: K = kc + {0..7,16..23} | {8..15,24..31}
  const int bklo = hl ? 8 : 0;      // B frag: K = kc + {0..7,16..23} | {8..15,24..31}

  const int brow = wave * 16 + ln;                      // batch row gathered by lane
  const int tok  = x[(size_t)brow * S_ + t];
  const unsigned short* arow = embb + (size_t)tok * E_; // emb row (bf16, 512B aligned)

  F8 c[4];
#pragma unroll
  for (int i = 0; i < 4; ++i)
#pragma unroll
    for (int j = 0; j < 8; ++j) c[i].f[j] = 0.0f;

#pragma unroll 2
  for (int kc = 0; kc < E_; kc += 32) {
    BF16Frag a;
    a.q[0] = *(const uint4*)(arow + kc + aklo);
    a.q[1] = *(const uint4*)(arow + kc + aklo + 16);
#pragma unroll
    for (int i = 0; i < 4; ++i) {
      const int n = n0 + i * 16 + ln;
      const unsigned short* bp = wxT + (size_t)n * E_ + kc + bklo;
      BF16Frag b;
      b.q[0] = *(const uint4*)(bp);
      b.q[1] = *(const uint4*)(bp + 16);
      c[i].v = __builtin_amdgcn_wmma_f32_16x16x32_bf16(
          false, a.v, false, b.v, (short)0, c[i].v, false, false);
    }
  }

#pragma unroll
  for (int i = 0; i < 4; ++i) {
    const int n = n0 + i * 16 + ln;
    const float bxv = bx[n];
#pragma unroll
    for (int j = 0; j < 8; ++j) {
      const int m = j + hl * 8;                          // row within 16x16 tile
      xw[((size_t)t * B_ + wave * 16 + m) * H_ + n] = f2bf(c[i].f[j] + bxv);
    }
  }
}

// ---------------- phase 2+3: persistent fused scan ------------------------------
// grid: B_/16 = 8 blocks; block: 256 = 8 waves. Each block owns a 16-row batch
// slice; h (16x512 bf16) lives in LDS; Wh^T / Wfc^T stream from L2 every step.
// The next step's xw tile is staged into an LDS double buffer by the Tensor Data
// Mover, overlapped with the current step's GEMM (TENSORcnt-tracked).
// Wave w: RNN-GEMM columns w*64..w*64+63 (4 tiles), FC column tile w*16..w*16+15.
__global__ __launch_bounds__(256) void rnn_scan_kernel(
    const unsigned short* __restrict__ xw,  const unsigned short* __restrict__ whT,
    const float* __restrict__ bh,           const unsigned short* __restrict__ wfcT,
    const float* __restrict__ bfc,          float* __restrict__ out)
{
  __shared__ __align__(16) unsigned short hsm[16 * HROW];
#if HAVE_TDM
  __shared__ __align__(16) unsigned short xwb[2][16 * HROW];
#endif

  const int b0   = blockIdx.x * 16;
  const int wave = threadIdx.x >> 5;
  const int lane = threadIdx.x & 31;
  const int hl   = lane >> 4;
  const int ln   = lane & 15;
  const int aklo = hl ? 8 : 0;
  const int bklo = hl ? 8 : 0;

  for (int i = threadIdx.x; i < 16 * HROW; i += 256) hsm[i] = 0;  // h0 = 0

  int ncol[4]; const unsigned short* whp[4]; float bhv[4];
#pragma unroll
  for (int i = 0; i < 4; ++i) {
    ncol[i] = wave * 64 + i * 16 + ln;
    whp[i]  = whT + (size_t)ncol[i] * H_;
    bhv[i]  = bh[ncol[i]];
  }
  const int nf = wave * 16 + ln;
  const unsigned short* wfp = wfcT + (size_t)nf * H_;
  const float bfcv = bfc[nf];

#if HAVE_TDM
  unsigned xwb_off[2];
  xwb_off[0] = (unsigned)(size_t)(void*)&xwb[0][0];   // LDS byte offset (flat low 32b)
  xwb_off[1] = (unsigned)(size_t)(void*)&xwb[1][0];
  if (wave == 0)  // prologue: stage xw tile for t = 0
    tdm_load_tile(xwb_off[0], xw + (size_t)b0 * H_);
#endif
  __syncthreads();

  for (int t = 0; t < S_; ++t) {
#if HAVE_TDM
    const int cur = t & 1;
    if (wave == 0) __builtin_amdgcn_s_wait_tensorcnt(0);  // tile t landed in LDS
    __syncthreads();
    if (wave == 0 && t + 1 < S_)  // stage tile t+1, overlapped with this step
      tdm_load_tile(xwb_off[1 - cur],
                    xw + (size_t)(t + 1) * B_ * H_ + (size_t)b0 * H_);
    const unsigned short* xb = &xwb[cur][0];
#endif

    // accumulators initialized with xw_t (bias already folded in by phase 1)
    F8 c[4];
#pragma unroll
    for (int i = 0; i < 4; ++i)
#pragma unroll
      for (int j = 0; j < 8; ++j) {
#if HAVE_TDM
        c[i].f[j] = bf2f(xb[(j + hl * 8) * HROW + ncol[i]]);
#else
        c[i].f[j] = bf2f(xw[(size_t)t * B_ * H_ +
                            (size_t)(b0 + j + hl * 8) * H_ + ncol[i]]);
#endif
      }

    // h_t = tanh(xw_t + h_{t-1} @ Wh + bh)
#pragma unroll 2
    for (int kc = 0; kc < H_; kc += 32) {
      BF16Frag a;
      const unsigned short* ap = &hsm[ln * HROW + kc + aklo];
      a.q[0] = *(const uint4*)(ap);
      a.q[1] = *(const uint4*)(ap + 16);
#pragma unroll
      for (int i = 0; i < 4; ++i) {
        const unsigned short* bp = whp[i] + kc + bklo;
        BF16Frag b;
        b.q[0] = *(const uint4*)(bp);
        b.q[1] = *(const uint4*)(bp + 16);
        c[i].v = __builtin_amdgcn_wmma_f32_16x16x32_bf16(
            false, a.v, false, b.v, (short)0, c[i].v, false, false);
      }
    }

    __syncthreads();   // all reads of h_{t-1} (this GEMM + prev step's FC) complete
#pragma unroll
    for (int i = 0; i < 4; ++i)
#pragma unroll
      for (int j = 0; j < 8; ++j) {
        const int m = j + hl * 8;
        hsm[m * HROW + ncol[i]] = f2bf(tanhf(c[i].f[j] + bhv[i]));
      }
    __syncthreads();   // h_t visible to all waves

    // fused FC: out[b, t, :] = h_t @ Wfc + bfc   (each wave: one 16-col tile)
    F8 oc;
#pragma unroll
    for (int j = 0; j < 8; ++j) oc.f[j] = 0.0f;
#pragma unroll 2
    for (int kc = 0; kc < H_; kc += 32) {
      BF16Frag a;
      const unsigned short* ap = &hsm[ln * HROW + kc + aklo];
      a.q[0] = *(const uint4*)(ap);
      a.q[1] = *(const uint4*)(ap + 16);
      const unsigned short* bp = wfp + kc + bklo;
      BF16Frag b;
      b.q[0] = *(const uint4*)(bp);
      b.q[1] = *(const uint4*)(bp + 16);
      oc.v = __builtin_amdgcn_wmma_f32_16x16x32_bf16(
          false, a.v, false, b.v, (short)0, oc.v, false, false);
    }
#pragma unroll
    for (int j = 0; j < 8; ++j) {
      const int m = j + hl * 8;
      out[(size_t)(b0 + m) * S_ * V_ + (size_t)t * V_ + nf] = oc.f[j] + bfcv;
    }
  }
}

// ---------------- launcher -------------------------------------------------------
extern "C" void kernel_launch(void* const* d_in, const int* in_sizes, int n_in,
                              void* d_out, int out_size, void* d_ws, size_t ws_size,
                              hipStream_t stream) {
  const int*   x   = (const int*)  d_in[0];
  const float* emb = (const float*)d_in[1];
  const float* Wx  = (const float*)d_in[2];
  const float* bx  = (const float*)d_in[3];
  const float* Wh  = (const float*)d_in[4];
  const float* bh  = (const float*)d_in[5];
  const float* Wfc = (const float*)d_in[6];
  const float* bfc = (const float*)d_in[7];
  float* out = (float*)d_out;

  // workspace layout (bytes):
  //   [0,       65536)  emb  bf16 (128x256)
  //   [65536,  327680)  Wx^T bf16 (512x256)
  //   [327680, 851968)  Wh^T bf16 (512x512)
  //   [851968, 983040)  Wfc^T bf16 (128x512)
  //   [1MiB,  ~257MiB)  xw   bf16 [S][B][H]
  char* ws = (char*)d_ws;
  unsigned short* embb = (unsigned short*)(ws + 0);
  unsigned short* wxT  = (unsigned short*)(ws + 65536);
  unsigned short* whT  = (unsigned short*)(ws + 327680);
  unsigned short* wfcT = (unsigned short*)(ws + 851968);
  unsigned short* xwp  = (unsigned short*)(ws + 1048576);

  cvt_bf16_kernel<<<(V_ * E_ + 255) / 256, 256, 0, stream>>>(emb, embb, V_ * E_);
  transpose_bf16_kernel<<<(E_ * H_ + 255) / 256, 256, 0, stream>>>(Wx,  wxT,  E_, H_);
  transpose_bf16_kernel<<<(H_ * H_ + 255) / 256, 256, 0, stream>>>(Wh,  whT,  H_, H_);
  transpose_bf16_kernel<<<(H_ * V_ + 255) / 256, 256, 0, stream>>>(Wfc, wfcT, H_, V_);

  rnn_xw_kernel<<<dim3(S_, H_ / 64), 256, 0, stream>>>(x, embb, wxT, bx, xwp);
  rnn_scan_kernel<<<B_ / 16, 256, 0, stream>>>(xwp, whT, bh, wfcT, bfc, out);
}
